// Transofrmer_38371237822506
// MI455X (gfx1250) — matrix operand order repaired and checked by reading, not compiled, not run on hardware
//
#include <hip/hip_runtime.h>

#define BB 16
#define RR 400
#define HH 8
#define QQ 64
#define DD 1200
#define DP 1216          // K padded to multiple of 32
#define HQ 512           // H*Q
#define MM (BB*RR)       // 6400 rows
#define MT (MM/16)       // 400 M tiles

typedef __attribute__((ext_vector_type(16))) __bf16        v16bf;
typedef __attribute__((ext_vector_type(8)))  float         v8f;
typedef __attribute__((ext_vector_type(8)))  unsigned int  v8u;

// round-to-nearest-even f32 -> bf16
static __device__ __forceinline__ unsigned short f2bf(float f) {
  union { float f; unsigned u; } v; v.f = f;
  unsigned r = v.u + 0x7FFFu + ((v.u >> 16) & 1u);
  return (unsigned short)(r >> 16);
}

// assemble a 16-element bf16 WMMA operand from two 16-byte chunks
static __device__ __forceinline__ v16bf mkv(uint4 a, uint4 b) {
  v8u t;
  t[0] = a.x; t[1] = a.y; t[2] = a.z; t[3] = a.w;
  t[4] = b.x; t[5] = b.y; t[6] = b.z; t[7] = b.w;
  return __builtin_bit_cast(v16bf, t);
}

// ---- stage 0: dtype conversion / padding / weight transpose -----------------

__global__ void conv_x_kernel(const float* __restrict__ x,
                              unsigned short* __restrict__ xb) {
  long long i = (long long)blockIdx.x * blockDim.x + threadIdx.x;
  if (i >= (long long)MM * DP) return;
  int r = (int)(i / DP), k = (int)(i % DP);
  float v = (k < DD) ? x[(long long)r * DD + k] : 0.0f;
  xb[i] = f2bf(v);
}

__global__ void conv_w_kernel(const float* __restrict__ w,
                              unsigned short* __restrict__ wt) {
  int i = blockIdx.x * blockDim.x + threadIdx.x;
  if (i >= HQ * DP) return;
  int n = i / DP, k = i % DP;
  float v = (k < DD) ? w[(long long)k * HQ + n] : 0.0f;   // transpose: wt[n][k]
  wt[i] = f2bf(v);
}

// ---- stage 1: Q/K projection GEMM (bf16 WMMA, f32 accum, +bias) -------------
// Block = one (matrix, M-tile): 8 waves x 4 N-tiles = full N=512.
// Software-pipelined K loop: loads for step k+1 issue before WMMAs of step k.

__global__ void __launch_bounds__(256) proj_gemm_kernel(
    const unsigned short* __restrict__ xb,
    const unsigned short* __restrict__ wqt,
    const unsigned short* __restrict__ wkt,
    const float* __restrict__ bq, const float* __restrict__ bk,
    unsigned short* __restrict__ qb, unsigned short* __restrict__ kb) {
  const int lane = threadIdx.x & 31;
  const int wave = threadIdx.x >> 5;
  const int mat  = (blockIdx.x >= MT) ? 1 : 0;
  const int mt   = mat ? (blockIdx.x - MT) : blockIdx.x;   // 0..399

  const unsigned short* wt  = mat ? wkt : wqt;
  const float*          bia = mat ? bk  : bq;
  unsigned short*       dst = mat ? kb  : qb;

  const int ha = (lane >> 4) << 3;             // accumulator half-wave row offset: 0 / 8
  const int hb = (lane >> 4) << 4;             // B half-wave K offset: 0 / 16
  const int l15 = lane & 15;

  const unsigned short* ap = xb + (long long)((mt << 4) + l15) * DP + hb;
  const int nt0 = wave << 2;                   // first of 4 N-tiles for this wave
  const unsigned short* bp0 = wt + (long long)(((nt0 + 0) << 4) + l15) * DP + hb;
  const unsigned short* bp1 = wt + (long long)(((nt0 + 1) << 4) + l15) * DP + hb;
  const unsigned short* bp2 = wt + (long long)(((nt0 + 2) << 4) + l15) * DP + hb;
  const unsigned short* bp3 = wt + (long long)(((nt0 + 3) << 4) + l15) * DP + hb;

  // pipeline prologue: operands for k0 = 0
  v16bf A  = mkv(*(const uint4*)(ap),       *(const uint4*)(ap + 8));
  v16bf B0 = mkv(*(const uint4*)(bp0),      *(const uint4*)(bp0 + 8));
  v16bf B1 = mkv(*(const uint4*)(bp1),      *(const uint4*)(bp1 + 8));
  v16bf B2 = mkv(*(const uint4*)(bp2),      *(const uint4*)(bp2 + 8));
  v16bf B3 = mkv(*(const uint4*)(bp3),      *(const uint4*)(bp3 + 8));

  v8f c0 = {}, c1 = {}, c2 = {}, c3 = {};
  for (int k0 = 0; k0 < DP; k0 += 32) {
    int kn = k0 + 32;
    if (kn >= DP) kn = 0;                      // branchless clamp (redundant last load)

    // speculative prefetch two K-steps ahead (dropped harmlessly at the tail)
    __builtin_prefetch(ap  + k0 + 64, 0, 1);
    __builtin_prefetch(bp0 + k0 + 64, 0, 1);
    __builtin_prefetch(bp2 + k0 + 64, 0, 1);

    // issue next-step loads before consuming current registers
    v16bf An  = mkv(*(const uint4*)(ap  + kn), *(const uint4*)(ap  + kn + 8));
    v16bf Bn0 = mkv(*(const uint4*)(bp0 + kn), *(const uint4*)(bp0 + kn + 8));
    v16bf Bn1 = mkv(*(const uint4*)(bp1 + kn), *(const uint4*)(bp1 + kn + 8));
    v16bf Bn2 = mkv(*(const uint4*)(bp2 + kn), *(const uint4*)(bp2 + kn + 8));
    v16bf Bn3 = mkv(*(const uint4*)(bp3 + kn), *(const uint4*)(bp3 + kn + 8));

    c0 = __builtin_amdgcn_wmma_f32_16x16x32_bf16(false, A, false, B0, (short)0, c0, false, false);
    c1 = __builtin_amdgcn_wmma_f32_16x16x32_bf16(false, A, false, B1, (short)0, c1, false, false);
    c2 = __builtin_amdgcn_wmma_f32_16x16x32_bf16(false, A, false, B2, (short)0, c2, false, false);
    c3 = __builtin_amdgcn_wmma_f32_16x16x32_bf16(false, A, false, B3, (short)0, c3, false, false);

    A = An; B0 = Bn0; B1 = Bn1; B2 = Bn2; B3 = Bn3;
  }

  // 16-row M tile never crosses a batch boundary (400 % 16 == 0):
  const int bi  = mt / 25;                     // wave-uniform -> SALU
  const int ri0 = (mt % 25) << 4;

  v8f cacc[4] = {c0, c1, c2, c3};
#pragma unroll
  for (int t = 0; t < 4; ++t) {
    const int n  = ((nt0 + t) << 4) + l15;
    const float bv = bia[n];
    const int h = n >> 6, qd = n & 63;
    long long base = (((long long)(bi * HH + h) * RR + (ri0 + ha)) << 6) + qd;
#pragma unroll
    for (int r = 0; r < 8; ++r)
      dst[base + ((long long)r << 6)] = f2bf(cacc[t][r] + bv);
  }
}

// ---- stage 2: fused QK^T / 64 -> softmax -> relu(. - thr) -------------------
// Block = (b, h, 16-row tile). 8 waves split 25 column tiles; K=64 via two
// chained 16x16x32 bf16 WMMAs. Pipelined B loads; softmax with shfl_xor.

__global__ void __launch_bounds__(256) attn_softmax_kernel(
    const unsigned short* __restrict__ qb,
    const unsigned short* __restrict__ kb,
    const float* __restrict__ thr_p,
    float* __restrict__ out) {
  __shared__ float srel[16 * RR];              // 25.6 KB

  const int lane = threadIdx.x & 31;
  const int wave = threadIdx.x >> 5;
  const int bh = blockIdx.x / 25;              // b*H + h
  const int it = blockIdx.x % 25;
  const int i0 = it << 4;
  const long long panel = (long long)bh * RR * QQ;

  const int ha = (lane >> 4) << 3;
  const int hb = (lane >> 4) << 4;
  const int l15 = lane & 15;

  // A operand: rows i0..i0+15, full K=64 (two 16x32 operands)
  const unsigned short* ap = qb + panel + (long long)(i0 + l15) * QQ;
  v16bf A0 = mkv(*(const uint4*)(ap + hb),      *(const uint4*)(ap + 8 + hb));
  v16bf A1 = mkv(*(const uint4*)(ap + 32 + hb), *(const uint4*)(ap + 40 + hb));

  const unsigned short* kpan = kb + panel + hb;

  // pipeline prologue for jt = wave
  {
    const unsigned short* bp = kpan + (long long)((wave << 4) + l15) * QQ;
    v16bf B0 = mkv(*(const uint4*)(bp),      *(const uint4*)(bp + 8));
    v16bf B1 = mkv(*(const uint4*)(bp + 32), *(const uint4*)(bp + 40));

    for (int jt = wave; jt < 25; jt += 8) {
      int jn = jt + 8;
      if (jn >= 25) jn = jt;                   // branchless clamp
      const unsigned short* bpn = kpan + (long long)((jn << 4) + l15) * QQ;
      v16bf Bn0 = mkv(*(const uint4*)(bpn),      *(const uint4*)(bpn + 8));
      v16bf Bn1 = mkv(*(const uint4*)(bpn + 32), *(const uint4*)(bpn + 40));

      v8f c = {};
      c = __builtin_amdgcn_wmma_f32_16x16x32_bf16(false, A0, false, B0, (short)0, c, false, false);
      c = __builtin_amdgcn_wmma_f32_16x16x32_bf16(false, A1, false, B1, (short)0, c, false, false);

      const int col = (jt << 4) + l15;
#pragma unroll
      for (int r = 0; r < 8; ++r)
        srel[(r + ha) * RR + col] = c[r] * (1.0f / 64.0f);   // /Q scaling

      B0 = Bn0; B1 = Bn1;
    }
  }
  __syncthreads();

  // softmax: 16 threads per row (rows are half-waves -> shfl width 16)
  const float thr = thr_p[0] * 0.01f;
  const int row = threadIdx.x >> 4;
  const int c0  = threadIdx.x & 15;
  float* rp = srel + row * RR;

  float mx = -3.402823466e38f;
#pragma unroll
  for (int col = 0; col < RR; col += 16) mx = fmaxf(mx, rp[col + c0]);
#pragma unroll
  for (int off = 8; off > 0; off >>= 1) mx = fmaxf(mx, __shfl_xor(mx, off, 16));

  float s = 0.0f;
#pragma unroll
  for (int col = 0; col < RR; col += 16) {
    float e = __expf(rp[col + c0] - mx);
    rp[col + c0] = e;
    s += e;
  }
#pragma unroll
  for (int off = 8; off > 0; off >>= 1) s += __shfl_xor(s, off, 16);
  const float inv = 1.0f / s;

  float* op = out + ((long long)bh * RR + (i0 + row)) * RR;
#pragma unroll
  for (int col = 0; col < RR; col += 16)
    op[col + c0] = fmaxf(rp[col + c0] * inv - thr, 0.0f);
}

// ---- launcher ---------------------------------------------------------------

extern "C" void kernel_launch(void* const* d_in, const int* in_sizes, int n_in,
                              void* d_out, int out_size, void* d_ws, size_t ws_size,
                              hipStream_t stream) {
  (void)in_sizes; (void)n_in; (void)out_size; (void)ws_size;
  const float* x  = (const float*)d_in[0];
  const float* Wq = (const float*)d_in[1];
  const float* bq = (const float*)d_in[2];
  const float* Wk = (const float*)d_in[3];
  const float* bk = (const float*)d_in[4];
  const float* th = (const float*)d_in[5];
  float* out = (float*)d_out;

  char* ws = (char*)d_ws;
  unsigned short* xb  = (unsigned short*)ws;  ws += (size_t)MM * DP * 2;   // 15.6 MB
  unsigned short* wqt = (unsigned short*)ws;  ws += (size_t)HQ * DP * 2;   // 1.25 MB
  unsigned short* wkt = (unsigned short*)ws;  ws += (size_t)HQ * DP * 2;   // 1.25 MB
  unsigned short* qb  = (unsigned short*)ws;  ws += (size_t)MM * HQ * 2;   // 6.55 MB
  unsigned short* kb  = (unsigned short*)ws;                               // 6.55 MB

  conv_x_kernel<<<(MM * DP + 255) / 256, 256, 0, stream>>>(x, xb);
  conv_w_kernel<<<(HQ * DP + 255) / 256, 256, 0, stream>>>(Wq, wqt);
  conv_w_kernel<<<(HQ * DP + 255) / 256, 256, 0, stream>>>(Wk, wkt);

  // 2 matrices * 400 M-tiles = 800 blocks; each block covers full N=512
  proj_gemm_kernel<<<2 * MT, 256, 0, stream>>>(xb, wqt, wkt, bq, bk, qb, kb);

  // 128 (b,h) panels * 25 row tiles = 3200 blocks
  attn_softmax_kernel<<<3200, 256, 0, stream>>>(qb, kb, th, out);
}
